// MambaBlock_46059229283209
// MI455X (gfx1250) — matrix operand order repaired and checked by reading, compiled, not run on hardware
//
#include <hip/hip_runtime.h>
#include <hip/hip_bf16.h>
#include <stdint.h>

typedef __bf16 bf16_t;
typedef bf16_t v8bf  __attribute__((ext_vector_type(8)));
typedef bf16_t v16bf __attribute__((ext_vector_type(16)));
typedef float  v8f   __attribute__((ext_vector_type(8)));
typedef int    v4i   __attribute__((ext_vector_type(4)));

#define LDSK 72   // padded LDS row stride in halves: 64 data + 8 pad = 144B (16B aligned)
#define TILEH (128 * LDSK)   // halves per 128x64 tile buffer

// ---- CDNA5 async global->LDS path (ASYNCcnt), compile-safe guard ----
#if defined(__has_builtin)
# if __has_builtin(__builtin_amdgcn_global_load_async_to_lds_b128) && \
     __has_builtin(__builtin_amdgcn_s_wait_asynccnt)
#  define HAVE_ASYNC_LDS 1
# endif
#endif
#ifndef HAVE_ASYNC_LDS
# define HAVE_ASYNC_LDS 0
#endif

#if HAVE_ASYNC_LDS
typedef __attribute__((address_space(1))) v4i* gv4i_t;   // global int4*
typedef __attribute__((address_space(3))) v4i* lv4i_t;   // LDS int4*
#define GLD_ASYNC_B128(gsrc, ldst) \
    __builtin_amdgcn_global_load_async_to_lds_b128((gv4i_t)(gsrc), (lv4i_t)(ldst), 0, 0)
#endif

// ---------------------------------------------------------------------------
// f32 -> bf16 pack, 8 elements per thread (b128 store)
// ---------------------------------------------------------------------------
__global__ __launch_bounds__(256)
void cvt_bf16_kernel(const float* __restrict__ in, bf16_t* __restrict__ out)
{
    const size_t i = (size_t)blockIdx.x * 256 + threadIdx.x;
    const float4* p = (const float4*)in + 2 * i;
    float4 a = p[0], b = p[1];
    v8bf o = { (bf16_t)a.x, (bf16_t)a.y, (bf16_t)a.z, (bf16_t)a.w,
               (bf16_t)b.x, (bf16_t)b.y, (bf16_t)b.z, (bf16_t)b.w };
    *((v8bf*)out + i) = o;
}

// ---------------------------------------------------------------------------
// C(MxN) = A(MxK,bf16) * W(NxK,bf16)^T + bias[N] (+ resid f32), f32 accum.
// Block tile 128x128, K-step 64 (2 WMMA sub-steps), 8 waves; wave = 4x2 tiles.
// Double-buffered LDS: async loads for tile k+1 overlap WMMAs on tile k;
// one barrier per K-tile.
// ---------------------------------------------------------------------------
template <bool RESID>
__global__ __launch_bounds__(256)
void wmma_gemm_bf16(const bf16_t* __restrict__ Amat,
                    const bf16_t* __restrict__ Wmat,
                    const float* __restrict__ bias,
                    const float* __restrict__ resid,
                    float* __restrict__ Cmat,
                    int M, int N, int K)
{
    __shared__ __attribute__((aligned(16))) bf16_t sA[2 * TILEH];
    __shared__ __attribute__((aligned(16))) bf16_t sB[2 * TILEH];

    const int tid  = threadIdx.x;
    const int lane = tid & 31;
    const int wave = tid >> 5;
    const int wm   = wave & 1;     // 0..1 : 64 M-rows per wave
    const int wn   = wave >> 1;    // 0..3 : 32 N-cols per wave
    const int bm   = blockIdx.y * 128;
    const int bn   = blockIdx.x * 128;

    // staging: thread moves 32 halves (64B) of A and of W per K-tile
    const int lrow = tid >> 1;           // 0..127
    const int lco  = (tid & 1) * 32;     // 0 or 32 halves

    const int lrowsel = lane & 15;
    const int hi      = (lane < 16) ? 0 : 1;

    const bf16_t* ga = Amat + (size_t)(bm + lrow) * K + lco;
    const bf16_t* gw = Wmat + (size_t)(bn + lrow) * K + lco;
    bf16_t* la = sA + lrow * LDSK + lco;
    bf16_t* lb = sB + lrow * LDSK + lco;

    v8f acc[4][2] = {};

#if HAVE_ASYNC_LDS
    // prologue: fill buffer 0 for kt = 0
    GLD_ASYNC_B128(ga +  0, la +  0);
    GLD_ASYNC_B128(ga +  8, la +  8);
    GLD_ASYNC_B128(ga + 16, la + 16);
    GLD_ASYNC_B128(ga + 24, la + 24);
    GLD_ASYNC_B128(gw +  0, lb +  0);
    GLD_ASYNC_B128(gw +  8, lb +  8);
    GLD_ASYNC_B128(gw + 16, lb + 16);
    GLD_ASYNC_B128(gw + 24, lb + 24);
#endif

    int buf = 0;
    for (int kt = 0; kt < K; kt += 64) {
#if HAVE_ASYNC_LDS
        // publish buffer `buf` (only its loads are outstanding), then kick off
        // the next tile into the other buffer and compute while it streams in.
        __builtin_amdgcn_s_wait_asynccnt(0);
        __syncthreads();
        if (kt + 64 < K) {
            const bf16_t* gan = ga + 64;
            const bf16_t* gwn = gw + 64;
            bf16_t* lan = la + (buf ? -TILEH : TILEH);
            bf16_t* lbn = lb + (buf ? -TILEH : TILEH);
            GLD_ASYNC_B128(gan +  0, lan +  0);
            GLD_ASYNC_B128(gan +  8, lan +  8);
            GLD_ASYNC_B128(gan + 16, lan + 16);
            GLD_ASYNC_B128(gan + 24, lan + 24);
            GLD_ASYNC_B128(gwn +  0, lbn +  0);
            GLD_ASYNC_B128(gwn +  8, lbn +  8);
            GLD_ASYNC_B128(gwn + 16, lbn + 16);
            GLD_ASYNC_B128(gwn + 24, lbn + 24);
            la = lan; lb = lbn;
            ga = gan; gw = gwn;
        }
#else
        v8bf t0 = *(const v8bf*)(ga + 0);
        v8bf t1 = *(const v8bf*)(ga + 8);
        v8bf t2 = *(const v8bf*)(ga + 16);
        v8bf t3 = *(const v8bf*)(ga + 24);
        v8bf u0 = *(const v8bf*)(gw + 0);
        v8bf u1 = *(const v8bf*)(gw + 8);
        v8bf u2 = *(const v8bf*)(gw + 16);
        v8bf u3 = *(const v8bf*)(gw + 24);
        if (kt + 64 < K) {
            __builtin_prefetch(ga + 64, 0, 1);   // global_prefetch_b8
            __builtin_prefetch(gw + 64, 0, 1);
        }
        bf16_t* lac = la + buf * TILEH;
        bf16_t* lbc = lb + buf * TILEH;
        *(v8bf*)(lac + 0)  = t0;
        *(v8bf*)(lac + 8)  = t1;
        *(v8bf*)(lac + 16) = t2;
        *(v8bf*)(lac + 24) = t3;
        *(v8bf*)(lbc + 0)  = u0;
        *(v8bf*)(lbc + 8)  = u1;
        *(v8bf*)(lbc + 16) = u2;
        *(v8bf*)(lbc + 24) = u3;
        __syncthreads();
        ga += 64; gw += 64;
#endif

        const bf16_t* cA = sA + buf * TILEH;
        const bf16_t* cB = sB + buf * TILEH;
        #pragma unroll
        for (int ks = 0; ks < 64; ks += 32) {
            v16bf af[4], bfr[2];
            #pragma unroll
            for (int mi = 0; mi < 4; ++mi) {
                // A 16x32: lanes<16 hold K 0-7 & 16-23; lanes>=16 hold K 8-15 & 24-31
                const bf16_t* pa = cA + (wm * 64 + mi * 16 + lrowsel) * LDSK + ks + hi * 8;
                v8bf lo  = *(const v8bf*)(pa);
                v8bf hi8 = *(const v8bf*)(pa + 16);
                af[mi] = __builtin_shufflevector(lo, hi8, 0,1,2,3,4,5,6,7,8,9,10,11,12,13,14,15);
            }
            #pragma unroll
            for (int ni = 0; ni < 2; ++ni) {
                // B 32x16: lanes<16 hold K 0-7 & 16-23; lanes>=16 hold K 8-15 & 24-31
                const bf16_t* pb = cB + (wn * 32 + ni * 16 + lrowsel) * LDSK + ks + hi * 8;
                v8bf lo  = *(const v8bf*)(pb);
                v8bf hi8 = *(const v8bf*)(pb + 16);
                bfr[ni] = __builtin_shufflevector(lo, hi8, 0,1,2,3,4,5,6,7,8,9,10,11,12,13,14,15);
            }
            #pragma unroll
            for (int mi = 0; mi < 4; ++mi)
                #pragma unroll
                for (int ni = 0; ni < 2; ++ni)
                    acc[mi][ni] = __builtin_amdgcn_wmma_f32_16x16x32_bf16(
                        false, af[mi], false, bfr[ni], (short)0, acc[mi][ni], false, false);
        }
        buf ^= 1;
    }

    // ---- epilogue: D layout = VGPR i: lanes<16 -> M=i, lanes>=16 -> M=8+i ----
    #pragma unroll
    for (int ni = 0; ni < 2; ++ni) {
        const int col = bn + wn * 32 + ni * 16 + lrowsel;
        const float bv = bias[col];
        #pragma unroll
        for (int mi = 0; mi < 4; ++mi) {
            const int row0 = bm + wm * 64 + mi * 16 + hi * 8;
            float* cp = Cmat + (size_t)row0 * N + col;
            const float* rp = RESID ? (resid + (size_t)row0 * N + col) : nullptr;
            #pragma unroll
            for (int i = 0; i < 8; ++i) {
                float v = acc[mi][ni][i] + bv;
                if (RESID) { v += *rp; rp += N; }
                *cp = v;
                cp += N;
            }
        }
    }
}

// ---------------------------------------------------------------------------
// delta[row,s] = softplus( dot(x_proj[row,:], B_w[s,:]) ), one wave per row.
// ---------------------------------------------------------------------------
__global__ __launch_bounds__(256)
void delta_kernel(const float* __restrict__ xp, const float* __restrict__ Bw,
                  float* __restrict__ delta)
{
    const int wave = threadIdx.x >> 5;
    const int lane = threadIdx.x & 31;
    const int row  = blockIdx.x * 8 + wave;
    const int s    = lane & 15;
    const int half = lane >> 4;

    const float4* xr = (const float4*)(xp + (size_t)row * 2048 + half * 1024);
    const float4* br = (const float4*)(Bw + (size_t)s * 2048 + half * 1024);
    float acc = 0.0f;
    #pragma unroll 4
    for (int i = 0; i < 256; ++i) {
        float4 xv = xr[i], bv = br[i];
        acc = fmaf(xv.x, bv.x, acc);
        acc = fmaf(xv.y, bv.y, acc);
        acc = fmaf(xv.z, bv.z, acc);
        acc = fmaf(xv.w, bv.w, acc);
    }
    acc += __shfl_xor(acc, 16, 32);
    if (lane < 16) {
        float z = acc;
        delta[(size_t)row * 16 + s] = fmaxf(z, 0.0f) + log1pf(__expf(-fabsf(z)));
    }
}

// ---------------------------------------------------------------------------
// Sequential scan. One block per (b,s); 128 threads own 16 channels each.
// ---------------------------------------------------------------------------
__global__ __launch_bounds__(128)
void scan_kernel(const float* __restrict__ xp, const float* __restrict__ delta,
                 const float* __restrict__ A, float* __restrict__ g)
{
    const int b   = blockIdx.x >> 4;
    const int s   = blockIdx.x & 15;
    const int tid = threadIdx.x;
    const int d0  = tid * 16;

    float h[16], a[16];
    #pragma unroll
    for (int j = 0; j < 16; ++j) {
        float av = A[(size_t)(d0 + j) * 16 + s];
        a[j] = 1.0f / (1.0f + __expf(-av));
        h[j] = 0.0f;
    }

    __shared__ float part[2][4];
    const float* xpb = xp + (size_t)b * 2048 * 2048 + d0;
    const float* dlb = delta + (size_t)b * 2048 * 16 + s;
    float*       gb  = g     + (size_t)b * 2048 * 16 + s;

    for (int t = 0; t < 2048; ++t) {
        const float dlt = dlb[(size_t)t * 16];
        const float4* xr = (const float4*)(xpb + (size_t)t * 2048);
        float partial = 0.0f;
        #pragma unroll
        for (int q = 0; q < 4; ++q) {
            float4 xv = xr[q];
            h[4*q+0] = fmaf(h[4*q+0], a[4*q+0], xv.x * dlt); partial += h[4*q+0];
            h[4*q+1] = fmaf(h[4*q+1], a[4*q+1], xv.y * dlt); partial += h[4*q+1];
            h[4*q+2] = fmaf(h[4*q+2], a[4*q+2], xv.z * dlt); partial += h[4*q+2];
            h[4*q+3] = fmaf(h[4*q+3], a[4*q+3], xv.w * dlt); partial += h[4*q+3];
        }
        #pragma unroll
        for (int off = 16; off > 0; off >>= 1)
            partial += __shfl_xor(partial, off, 32);
        if ((tid & 31) == 0) part[t & 1][tid >> 5] = partial;
        __syncthreads();
        if (tid == 0)
            gb[(size_t)t * 16] = part[t & 1][0] + part[t & 1][1]
                               + part[t & 1][2] + part[t & 1][3];
    }
}

// ---------------------------------------------------------------------------
// ymid[row,d] = dot16(g[row,:], C_w[d,:]) + D_param[d]*x_proj[row,d]  -> bf16
// Thread handles 8 consecutive d, stores one b128 of bf16.
// ---------------------------------------------------------------------------
__global__ __launch_bounds__(256)
void ycombine_kernel(const float* __restrict__ g, const float* __restrict__ Cw,
                     const float* __restrict__ Dp, const float* __restrict__ xp,
                     bf16_t* __restrict__ ybf)
{
    const size_t gid = (size_t)blockIdx.x * 256 + threadIdx.x;  // over M * D/8
    const int row = (int)(gid >> 8);
    const int d0  = (int)(gid & 255) * 8;

    const float4* gr = (const float4*)(g + (size_t)row * 16);
    float4 g0 = gr[0], g1 = gr[1], g2 = gr[2], g3 = gr[3];
    const float4* xr = (const float4*)(xp + (size_t)row * 2048 + d0);
    float4 xv0 = xr[0], xv1 = xr[1];
    const float4* dr = (const float4*)(Dp + d0);
    float4 dp0 = dr[0], dp1 = dr[1];
    const float xs[8] = { xv0.x, xv0.y, xv0.z, xv0.w, xv1.x, xv1.y, xv1.z, xv1.w };
    const float ds[8] = { dp0.x, dp0.y, dp0.z, dp0.w, dp1.x, dp1.y, dp1.z, dp1.w };

    float out[8];
    #pragma unroll
    for (int j = 0; j < 8; ++j) {
        const float4* cr = (const float4*)(Cw + (size_t)(d0 + j) * 16);
        float4 c0 = cr[0], c1 = cr[1], c2 = cr[2], c3 = cr[3];
        float acc = 0.0f;
        acc = fmaf(g0.x, c0.x, acc); acc = fmaf(g0.y, c0.y, acc);
        acc = fmaf(g0.z, c0.z, acc); acc = fmaf(g0.w, c0.w, acc);
        acc = fmaf(g1.x, c1.x, acc); acc = fmaf(g1.y, c1.y, acc);
        acc = fmaf(g1.z, c1.z, acc); acc = fmaf(g1.w, c1.w, acc);
        acc = fmaf(g2.x, c2.x, acc); acc = fmaf(g2.y, c2.y, acc);
        acc = fmaf(g2.z, c2.z, acc); acc = fmaf(g2.w, c2.w, acc);
        acc = fmaf(g3.x, c3.x, acc); acc = fmaf(g3.y, c3.y, acc);
        acc = fmaf(g3.z, c3.z, acc); acc = fmaf(g3.w, c3.w, acc);
        out[j] = fmaf(ds[j], xs[j], acc);
    }
    v8bf o = { (bf16_t)out[0], (bf16_t)out[1], (bf16_t)out[2], (bf16_t)out[3],
               (bf16_t)out[4], (bf16_t)out[5], (bf16_t)out[6], (bf16_t)out[7] };
    *(v8bf*)(ybf + gid * 8) = o;
}

// ---------------------------------------------------------------------------
extern "C" void kernel_launch(void* const* d_in, const int* in_sizes, int n_in,
                              void* d_out, int out_size, void* d_ws, size_t ws_size,
                              hipStream_t stream)
{
    const float* x    = (const float*)d_in[0];
    const float* Win  = (const float*)d_in[1];
    const float* bin  = (const float*)d_in[2];
    const float* Wout = (const float*)d_in[3];
    const float* bout = (const float*)d_in[4];
    const float* A    = (const float*)d_in[5];
    const float* Bw   = (const float*)d_in[6];
    const float* Cw   = (const float*)d_in[7];
    const float* Dp   = (const float*)d_in[8];
    float* out = (float*)d_out;

    const int Mrows = 4 * 2048;   // 8192
    const int D = 2048;
    const size_t matF32 = (size_t)Mrows * D * sizeof(float);   // 64 MB
    const size_t matBF  = (size_t)Mrows * D * sizeof(bf16_t);  // 32 MB
    const size_t wBF    = (size_t)D * D * sizeof(bf16_t);      //  8 MB
    const size_t gsF32  = (size_t)Mrows * 16 * sizeof(float);  // 512 KB

    char* ws = (char*)d_ws;
    float*  xp     = (float*)(ws);                                   // 64 MB
    bf16_t* xbf    = (bf16_t*)(ws + matF32);                         // 32 MB
    bf16_t* ybf    = (bf16_t*)(ws + matF32 + matBF);                 // 32 MB
    bf16_t* winbf  = (bf16_t*)(ws + matF32 + 2 * matBF);             //  8 MB
    bf16_t* woutbf = (bf16_t*)(ws + matF32 + 2 * matBF + wBF);       //  8 MB
    float*  delta  = (float*)(ws + matF32 + 2 * matBF + 2 * wBF);    // 512 KB
    float*  g      = (float*)(ws + matF32 + 2 * matBF + 2 * wBF + gsF32);

    dim3 gg(D / 128, Mrows / 128);   // (16, 64)

    // 0) pack operands to bf16 once (halves GEMM HBM traffic)
    cvt_bf16_kernel<<<(Mrows * D) / (256 * 8), 256, 0, stream>>>(x, xbf);
    cvt_bf16_kernel<<<(D * D) / (256 * 8), 256, 0, stream>>>(Win, winbf);
    cvt_bf16_kernel<<<(D * D) / (256 * 8), 256, 0, stream>>>(Wout, woutbf);
    // 1) x_proj = x @ W_in^T + b_in
    wmma_gemm_bf16<false><<<gg, 256, 0, stream>>>(xbf, winbf, bin, nullptr, xp, Mrows, D, D);
    // 2) delta = softplus(x_proj @ B_w^T)
    delta_kernel<<<Mrows / 8, 256, 0, stream>>>(xp, Bw, delta);
    // 3) sequential scan -> g[b,t,s]
    scan_kernel<<<4 * 16, 128, 0, stream>>>(xp, delta, A, g);
    // 4) y = g @ C_w^T + D ⊙ x_proj   (bf16 output feeds GEMM2 directly)
    ycombine_kernel<<<(Mrows * D) / (256 * 8), 256, 0, stream>>>(g, Cw, Dp, xp, ybf);
    // 5) out = y @ W_out^T + b_out + x
    wmma_gemm_bf16<true><<<gg, 256, 0, stream>>>(ybf, woutbf, bout, x, out, Mrows, D, D);
}
